// LTM_predictor_11493332484862
// MI455X (gfx1250) — matrix operand order repaired and checked by reading, not compiled, not run on hardware
//
#include <hip/hip_runtime.h>
#include <hip/hip_fp16.h>

#define Hc   512
#define Wcc  512
#define HWc  (512 * 512)
#define Bc   8
#define Ttc  16
#define NCH  10

typedef __attribute__((ext_vector_type(2)))  float    v2f;
typedef __attribute__((ext_vector_type(8)))  float    v8f;
typedef __attribute__((ext_vector_type(16))) _Float16 v16h;

// ---------------------------------------------------------------------------
// gfx1250 async global->LDS copy (ASYNCcnt-tracked, VGPR-bypassing) with a
// portable fallback. Guarded so compilation never breaks.
// ---------------------------------------------------------------------------
#if __has_builtin(__builtin_amdgcn_global_load_async_to_lds_b32)
#define ASYNC_LDS 1
typedef __attribute__((address_space(1))) int* gint_p;
typedef __attribute__((address_space(3))) int* lint_p;
#else
#define ASYNC_LDS 0
#endif

__device__ __forceinline__ void async_copy_b32(const float* g, float* l)
{
#if ASYNC_LDS
    __builtin_amdgcn_global_load_async_to_lds_b32((gint_p)g, (lint_p)l, 0, 0);
#else
    *l = *g;
#endif
}

__device__ __forceinline__ void async_wait_all()
{
#if ASYNC_LDS
#if __has_builtin(__builtin_amdgcn_s_wait_asynccnt)
    __builtin_amdgcn_s_wait_asynccnt(0);
#else
    asm volatile("s_wait_asynccnt 0x0" ::: "memory");
#endif
#endif
}

// ---------------------------------------------------------------------------
// Conv: T[b, o, i, j] = bias[o] + sum_{c,kh,kw} W[o,c,kh,kw] * x[b, 12+c, 0, i+kh-1, j+kw-1]
// GEMM formulation: M=16 (10 out ch + pad), K=36 (4ch x 9 taps), N=16 pixels/wave.
// V_WMMA_F32_16X16X4_F32, 9 chained K=4 steps (exact fp32). All loads are
// unconditional (clamped address + select-0) so EXEC stays all-ones — required
// by WMMA and avoids s_and_saveexec churn.
// ---------------------------------------------------------------------------
__global__ __launch_bounds__(256)
void conv_T_wmma(const float* __restrict__ x, const float* __restrict__ Wt,
                 const float* __restrict__ bias, float* __restrict__ T)
{
    int gtid = blockIdx.x * 256 + threadIdx.x;
    int wave = gtid >> 5;
    int lane = threadIdx.x & 31;
    int half = lane >> 4;   // 0: lanes 0-15, 1: lanes 16-31
    int ln   = lane & 15;   // M (A rows) / N (B cols)

    const int SPR = Wcc / 16;     // 32 strips per image row
    const int SPB = Hc * SPR;     // strips per batch
    int b   = wave / SPB;
    int rem = wave - b * SPB;
    int i   = rem / SPR;
    int j0  = (rem - i * SPR) * 16;

    // C init = bias broadcast; C/D layout: vgpr r -> M = r + 8*half, N = ln
    v8f acc;
#pragma unroll
    for (int r = 0; r < 8; ++r) {
        int M  = r + 8 * half;
        float bv = bias[M < NCH ? M : (NCH - 1)];    // unconditional load
        acc[r] = (M < NCH) ? bv : 0.0f;
    }

    const float* xb = x + (size_t)b * (Ttc * HWc);

#if __has_builtin(__builtin_amdgcn_wmma_f32_16x16x4_f32)
#pragma unroll
    for (int s = 0; s < 9; ++s) {
        // f32 16x16x4 fragment layout: lanes 0-15 carry K = 4s+{0,1} in v0/v1,
        // lanes 16-31 carry K = 4s+{2,3}.
        v2f a, bm;
#pragma unroll
        for (int q = 0; q < 2; ++q) {
            int k = 4 * s + 2 * half + q;            // 0..35
            float av = Wt[(ln < NCH ? ln : (NCH - 1)) * 36 + k];
            a[q] = (ln < NCH) ? av : 0.0f;

            int c  = k / 9;
            int t9 = k - 9 * c;
            int kh = t9 / 3;
            int kw = t9 - 3 * kh;
            int gi = i + kh - 1;
            int gj = j0 + ln + kw - 1;
            bool ok = ((unsigned)gi < (unsigned)Hc) & ((unsigned)gj < (unsigned)Wcc);
            int cgi = gi < 0 ? 0 : (gi > Hc - 1 ? Hc - 1 : gi);
            int cgj = gj < 0 ? 0 : (gj > Wcc - 1 ? Wcc - 1 : gj);
            float v = xb[(size_t)(12 + c) * HWc + (size_t)cgi * Wcc + cgj];
            bm[q] = ok ? v : 0.0f;
        }
        acc = __builtin_amdgcn_wmma_f32_16x16x4_f32(
            false, a, false, bm, (short)0, acc, false, false);
    }
#else
    // Fallback: two f16 16x16x32 WMMAs, K zero-padded 36 -> 64.
#pragma unroll
    for (int m = 0; m < 2; ++m) {
        v16h a, bm;
#pragma unroll
        for (int e = 0; e < 16; ++e) {
            int p = e >> 1, q = e & 1;
            int ka = ((p & 3) * 2) + q + ((p >> 2) * 16) + (half ? 8 : 0) + m * 32;
            float av = 0.0f;
            if (ka < 36 && ln < NCH) av = Wt[ln * 36 + ka];
            a[e] = (_Float16)av;
            int kb = ((p & 3) * 2) + q + ((p >> 2) * 16) + (half ? 8 : 0) + m * 32;
            float bv = 0.0f;
            if (kb < 36) {
                int c  = kb / 9;
                int t9 = kb - 9 * c;
                int kh = t9 / 3;
                int kw = t9 - 3 * kh;
                int gi = i + kh - 1;
                int gj = j0 + ln + kw - 1;
                if ((unsigned)gi < (unsigned)Hc && (unsigned)gj < (unsigned)Wcc)
                    bv = xb[(size_t)(12 + c) * HWc + (size_t)gi * Wcc + gj];
            }
            bm[e] = (_Float16)bv;
        }
        acc = __builtin_amdgcn_wmma_f32_16x16x32_f16(
            false, a, false, bm, (short)0, acc, false, false);
    }
#endif

    // Store rows M < 10 (necessary divergence; after all WMMAs)
    float* Tb = T + (size_t)b * NCH * HWc + (size_t)i * Wcc + j0 + ln;
#pragma unroll
    for (int r = 0; r < 8; ++r) {
        int M = r + 8 * half;
        if (M < NCH) Tb[(size_t)M * HWc] = acc[r];
    }
}

// ---------------------------------------------------------------------------
// First LTM step: xt0 = ltm(x[:, 15, 0], T); writes ping buffer + d_out[t=0].
// 32x8 tile + 1-halo staged in LDS via async global->LDS copies.
// ---------------------------------------------------------------------------
__global__ __launch_bounds__(256)
void ltm_first(const float* __restrict__ x, const float* __restrict__ T,
               float* __restrict__ xout, float* __restrict__ dout)
{
    __shared__ float xs[10][34];
    int tx = threadIdx.x, ty = threadIdx.y;
    int tid = ty * 32 + tx;
    int i0 = blockIdx.y * 8, j0 = blockIdx.x * 32, b = blockIdx.z;

    const float* xb = x + ((size_t)b * Ttc + 15) * HWc;
    for (int l = tid; l < 340; l += 256) {
        int r = l / 34, c = l - 34 * r;
        int gi = i0 + r - 1, gj = j0 + c - 1;
        if ((unsigned)gi < (unsigned)Hc && (unsigned)gj < (unsigned)Wcc)
            async_copy_b32(xb + (size_t)gi * Wcc + gj, &xs[r][c]);
        else
            xs[r][c] = 0.0f;
    }
    async_wait_all();
    __syncthreads();

    int gi = i0 + ty, gj = j0 + tx;
    const float* Tp = T + (size_t)b * NCH * HWc + (size_t)gi * Wcc + gj;
    float out = Tp[(size_t)9 * HWc];
    int k = 0;
#pragma unroll
    for (int dj = 0; dj < 3; ++dj)
#pragma unroll
        for (int di = 0; di < 3; ++di) {
            out += xs[ty + di][tx + dj] * Tp[(size_t)k * HWc];
            ++k;
        }
    xout[(size_t)b * HWc + (size_t)gi * Wcc + gj] = out;
    dout[((size_t)b * Ttc + 0) * HWc + (size_t)gi * Wcc + gj] = out;
}

// ---------------------------------------------------------------------------
// Fused pair: out = sigmoid(ltm(ltm(x))) — matches the reference inner pattern.
// 2-deep halo: xs 12x36 (async-filled), intermediate ms 10x34 in LDS.
// Halves global syncs and x traffic vs. unfused; T stays resident in L2.
// ---------------------------------------------------------------------------
__global__ __launch_bounds__(256)
void ltm_pair(const float* __restrict__ xin, const float* __restrict__ T,
              float* __restrict__ xout, float* __restrict__ dout, int out_t)
{
    __shared__ float xs[12][36];
    __shared__ float ms[10][34];
    int tx = threadIdx.x, ty = threadIdx.y;
    int tid = ty * 32 + tx;
    int i0 = blockIdx.y * 8, j0 = blockIdx.x * 32, b = blockIdx.z;

    const float* xb = xin + (size_t)b * HWc;
    const float* Tb = T + (size_t)b * NCH * HWc;

    for (int l = tid; l < 432; l += 256) {
        int r = l / 36, c = l - 36 * r;
        int gi = i0 + r - 2, gj = j0 + c - 2;
        if ((unsigned)gi < (unsigned)Hc && (unsigned)gj < (unsigned)Wcc)
            async_copy_b32(xb + (size_t)gi * Wcc + gj, &xs[r][c]);
        else
            xs[r][c] = 0.0f;
    }
    async_wait_all();
    __syncthreads();

    for (int l = tid; l < 340; l += 256) {
        int r = l / 34, c = l - 34 * r;
        int gi = i0 + r - 1, gj = j0 + c - 1;
        float v = 0.0f;
        if ((unsigned)gi < (unsigned)Hc && (unsigned)gj < (unsigned)Wcc) {
            const float* Tp = Tb + (size_t)gi * Wcc + gj;
            v = Tp[(size_t)9 * HWc];
            int k = 0;
#pragma unroll
            for (int dj = 0; dj < 3; ++dj)
#pragma unroll
                for (int di = 0; di < 3; ++di) {
                    v += xs[r + di][c + dj] * Tp[(size_t)k * HWc];
                    ++k;
                }
        }
        ms[r][c] = v;
    }
    __syncthreads();

    int gi = i0 + ty, gj = j0 + tx;
    const float* Tp = Tb + (size_t)gi * Wcc + gj;
    float out = Tp[(size_t)9 * HWc];
    int k = 0;
#pragma unroll
    for (int dj = 0; dj < 3; ++dj)
#pragma unroll
        for (int di = 0; di < 3; ++di) {
            out += ms[ty + di][tx + dj] * Tp[(size_t)k * HWc];
            ++k;
        }
    out = 1.0f / (1.0f + __expf(-out));
    xout[(size_t)b * HWc + (size_t)gi * Wcc + gj] = out;
    if (out_t >= 0)
        dout[((size_t)b * Ttc + out_t) * HWc + (size_t)gi * Wcc + gj] = out;
}

// ---------------------------------------------------------------------------
extern "C" void kernel_launch(void* const* d_in, const int* in_sizes, int n_in,
                              void* d_out, int out_size, void* d_ws, size_t ws_size,
                              hipStream_t stream)
{
    (void)in_sizes; (void)n_in; (void)out_size; (void)ws_size;
    const float* x    = (const float*)d_in[0];
    const float* Wt   = (const float*)d_in[1];
    const float* bias = (const float*)d_in[2];
    float* out = (float*)d_out;
    float* ws  = (float*)d_ws;

    float* T   = ws;                                   // 8*10*512*512 floats (83.9 MB)
    float* xb0 = ws + (size_t)Bc * NCH * HWc;          // ping (8.4 MB)
    float* xb1 = xb0 + (size_t)Bc * HWc;               // pong (8.4 MB)

    // 8*512*32 strips * 32 lanes / 256 threads = 16384 blocks (exact waves -> EXEC all-ones)
    conv_T_wmma<<<16384, 256, 0, stream>>>(x, Wt, bias, T);

    dim3 blk(32, 8);
    dim3 grd(Wcc / 32, Hc / 8, Bc);

    ltm_first<<<grd, blk, 0, stream>>>(x, T, xb0, out);

    float* cur = xb0;
    float* nxt = xb1;
    for (int it = 1; it < Ttc; ++it) {
        for (int j = 0; j < 3; ++j) {
            int ot = (j == 2) ? it : -1;
            ltm_pair<<<grd, blk, 0, stream>>>(cur, T, nxt, out, ot);
            float* tmp = cur; cur = nxt; nxt = tmp;
        }
    }
}
